// E63bIndependentGatesCell_4887672783119
// MI455X (gfx1250) — matrix operand order repaired and checked by reading, not compiled, not run on hardware
//
#include <hip/hip_runtime.h>

#define Dd 1024
#define Bb 16

typedef __attribute__((ext_vector_type(16))) __bf16 v16bf;
typedef __attribute__((ext_vector_type(8))) float v8f;
typedef __attribute__((ext_vector_type(4))) unsigned int u32x4;
typedef __attribute__((ext_vector_type(8))) int i32x8;
typedef __attribute__((ext_vector_type(4))) int i32x4;

union BFrag { v16bf v; unsigned short s[16]; ushort4 q[4]; };

__device__ __forceinline__ unsigned short f2bf(float f){
  union { float f; unsigned u; } c; c.f = f;
  unsigned u = c.u + 0x7FFFu + ((c.u >> 16) & 1u);   // round-to-nearest-even
  return (unsigned short)(u >> 16);
}

__device__ __forceinline__ v8f wmma_bf16(v16bf a, v16bf b, v8f c){
  // D = A(16x32 bf16) x B(32x16 bf16) + C(16x16 f32)
  return __builtin_amdgcn_wmma_f32_16x16x32_bf16(false, a, false, b, (short)0, c, false, false);
}

// A fragment (16x32 bf16) from LDS tile [.][32], rows rowbase..rowbase+15.
// ISA layout: lane<16 -> M=lane, elems = K0..7,K16..23 ; lane>=16 -> M=lane-16, K8..15,K24..31
__device__ __forceinline__ v16bf make_afrag(const unsigned short (*At)[32], int rowbase){
  int lane = threadIdx.x & 31;
  int row  = rowbase + (lane & 15);
  int k0   = (lane >> 4) ? 8 : 0;
  BFrag u;
  const unsigned short* p = &At[row][0];
  u.q[0] = *(const ushort4*)&p[k0 + 0];
  u.q[1] = *(const ushort4*)&p[k0 + 4];
  u.q[2] = *(const ushort4*)&p[k0 + 16];
  u.q[3] = *(const ushort4*)&p[k0 + 20];
  return u.v;
}

// B fragment (32x16 bf16), LDS stored [n][k].
// ISA layout: lane<16 -> N=lane, K0..7,K16..23 ; lane>=16 -> N=lane-16, K8..15,K24..31
__device__ __forceinline__ v16bf make_bfrag(const unsigned short (*Bt)[32], int nbase){
  int lane = threadIdx.x & 31;
  int col  = nbase + (lane & 15);
  int ks   = (lane >> 4) ? 8 : 0;
  BFrag u;
  const unsigned short* p = &Bt[col][ks];
  u.q[0] = *(const ushort4*)&p[0];
  u.q[1] = *(const ushort4*)&p[4];
  u.q[2] = *(const ushort4*)&p[16];
  u.q[3] = *(const ushort4*)&p[20];
  return u.v;
}

// ---------------------------------------------------------------------------
// Phase 1: fused projection GEMMs. out = sigmoid/identity(X @ W^T + bias) for
// W_alpha, W_beta, W_x sharing one pass over X. Block tile 128(M) x 64(N),
// K-chunks of 32, 8 waves = 4x2 wave grid, each wave 32x32 (2x2 WMMA tiles)
// per weight.
// ---------------------------------------------------------------------------
__global__ __launch_bounds__(256) void proj_gemm(
    const float* __restrict__ x,
    const float* __restrict__ Wa, const float* __restrict__ Wb, const float* __restrict__ Wx,
    const float* __restrict__ ba, const float* __restrict__ bb, const float* __restrict__ bx,
    float* __restrict__ alpha, float* __restrict__ beta, float* __restrict__ wxo)
{
  __shared__ alignas(32) unsigned short At[128][32];     // 8 KB
  __shared__ alignas(32) unsigned short Bt[3][64][32];   // 12 KB
  const int tid  = threadIdx.x;
  const int lane = tid & 31;
  const int w    = tid >> 5;
  const int wm   = w & 3;
  const int wn   = w >> 2;
  const int m0   = blockIdx.y * 128;
  const int n0   = blockIdx.x * 64;
  const float* Ws[3] = {Wa, Wb, Wx};

  v8f acc[3][2][2] = {};

  for (int k0 = 0; k0 < Dd; k0 += 32){
    #pragma unroll
    for (int e = 0; e < 4; ++e){                 // A tile: 128x32 f32 -> bf16
      int i4 = e*256 + tid;                      // float4 index, 1024 total
      int r = i4 >> 3, c4 = i4 & 7;
      float4 f = *(const float4*)&x[(size_t)(m0 + r)*Dd + k0 + c4*4];
      unsigned short* d = &At[r][c4*4];
      d[0]=f2bf(f.x); d[1]=f2bf(f.y); d[2]=f2bf(f.z); d[3]=f2bf(f.w);
    }
    #pragma unroll
    for (int g = 0; g < 3; ++g){                 // B tiles: W rows are contiguous in k
      #pragma unroll
      for (int e = 0; e < 2; ++e){
        int i4 = e*256 + tid;                    // 512 float4s
        int n = i4 >> 3, c4 = i4 & 7;
        float4 f = *(const float4*)&Ws[g][(size_t)(n0 + n)*Dd + k0 + c4*4];
        unsigned short* d = &Bt[g][n][c4*4];
        d[0]=f2bf(f.x); d[1]=f2bf(f.y); d[2]=f2bf(f.z); d[3]=f2bf(f.w);
      }
    }
    __syncthreads();

    v16bf a0 = make_afrag(At, wm*32);
    v16bf a1 = make_afrag(At, wm*32 + 16);
    #pragma unroll
    for (int g = 0; g < 3; ++g){
      v16bf b0 = make_bfrag(Bt[g], wn*32);
      v16bf b1 = make_bfrag(Bt[g], wn*32 + 16);
      acc[g][0][0] = wmma_bf16(a0, b0, acc[g][0][0]);
      acc[g][0][1] = wmma_bf16(a0, b1, acc[g][0][1]);
      acc[g][1][0] = wmma_bf16(a1, b0, acc[g][1][0]);
      acc[g][1][1] = wmma_bf16(a1, b1, acc[g][1][1]);
    }
    __syncthreads();
  }

  // Epilogue: C/D layout -> VGPR r holds row (r + 8*hi), col = lane&15
  const int hi = lane >> 4;
  const int ln = lane & 15;
  float* outs[3] = {alpha, beta, wxo};
  const float* biases[3] = {ba, bb, bx};
  #pragma unroll
  for (int g = 0; g < 3; ++g){
    #pragma unroll
    for (int ni = 0; ni < 2; ++ni){
      int nn = n0 + wn*32 + ni*16 + ln;
      float bias = biases[g][nn];
      #pragma unroll
      for (int mi = 0; mi < 2; ++mi){
        #pragma unroll
        for (int r = 0; r < 8; ++r){
          int mm = m0 + wm*32 + mi*16 + hi*8 + r;
          float v = acc[g][mi][ni][r] + bias;
          if (g < 2) v = 1.0f / (1.0f + __expf(-v));   // sigmoid for alpha/beta
          outs[g][(size_t)mm*Dd + nn] = v;
        }
      }
    }
  }
}

// ---------------------------------------------------------------------------
// Phase 2: persistent recurrent scan. 64 blocks x 128 threads; block b owns
// output columns [16b, 16b+16). W_h slice pinned in LDS as bf16. Each step:
// TDM tensor_load of h[t] (16x1024 f32, 64KB) into LDS, 16x16 WMMA tile with
// K=1024 split over 4 waves, LDS reduce, gate epilogue by wave 0, grid
// barrier.
// ---------------------------------------------------------------------------
__global__ __launch_bounds__(128) void rnn_scan(
    const float* __restrict__ Wh,
    const float* __restrict__ alpha, const float* __restrict__ beta,
    const float* __restrict__ wx,
    float* __restrict__ out, float* __restrict__ h,
    unsigned* bar_cnt, unsigned* bar_gen, int T, int nblk)
{
  __shared__ alignas(32) unsigned short wlds[16][1024];  // 32 KB  B-matrix [n][k] bf16
  __shared__ alignas(32) float hst[16][1024];            // 64 KB  staged h[t] f32
  __shared__ alignas(32) float red[3][256];              //  3 KB  wave partials

  const int tid  = threadIdx.x;
  const int lane = tid & 31;
  const int w    = tid >> 5;
  const int hi   = lane >> 4;
  const int ln   = lane & 15;
  const int N0   = blockIdx.x * 16;
  const int BD   = Bb * Dd;

  // Load this block's W_h^T slice once: B[k][n] = Wh[N0+n][k] (rows contiguous in k)
  #pragma unroll
  for (int i = 0; i < 32; ++i){
    int i4 = i*128 + tid;                 // 4096 float4s
    int n = i4 >> 8, c4 = i4 & 255;
    float4 f = *(const float4*)&Wh[(size_t)(N0 + n)*Dd + c4*4];
    unsigned short* d = &wlds[n][c4*4];
    d[0]=f2bf(f.x); d[1]=f2bf(f.y); d[2]=f2bf(f.z); d[3]=f2bf(f.w);
  }
  __syncthreads();

  // Low 32 bits of a generic LDS pointer == LDS byte offset (aperture rule).
  const unsigned hst_lds = (unsigned)(unsigned long long)&hst[0][0];

  for (int t = 0; t < T; ++t){
    // --- stage h[t] into LDS via one Tensor Data Mover descriptor -----------
    if (w == 0){
      unsigned long long gaddr = (unsigned long long)(h + (size_t)t * BD);
      // D# group 0: count=1, lds_addr[63:32], global_addr[120:64], type=2
      u32x4 g0;
      g0.x = 1u;
      g0.y = hst_lds;
      g0.z = (unsigned)gaddr;
      g0.w = (unsigned)((gaddr >> 32) & 0x01FFFFFFu) | (2u << 30);
      // D# group 1: data_size=4B(2); tensor_dim0=1024, tensor_dim1=16;
      // tile_dim0=1024, tile_dim1=16; tensor_dim0_stride=1024
      i32x8 g1 = (i32x8)0;
      g1[0] = (int)(2u << 16);            // data_size=2 (4 bytes)
      g1[1] = (int)(1024u << 16);         // tensor_dim0[15:0] at bits 63:48
      g1[2] = (int)(16u << 16);           // tensor_dim0[31:16]=0 | tensor_dim1[15:0]
      g1[3] = (int)(1024u << 16);         // tensor_dim1[31:16]=0 | tile_dim0=1024
      g1[4] = 16;                         // tile_dim1=16, tile_dim2=0
      g1[5] = 1024;                       // tensor_dim0_stride[31:0]
      i32x4 gz = (i32x4)0;                // groups 2/3 unused (2-D tensor)
#if __clang_major__ >= 23
      i32x8 gz8 = (i32x8)0;
      __builtin_amdgcn_tensor_load_to_lds(g0, g1, gz, gz, gz8, 0);
#else
      __builtin_amdgcn_tensor_load_to_lds(g0, g1, gz, gz, 0);
#endif
      __builtin_amdgcn_s_wait_tensorcnt(0);
    }
    __syncthreads();

    // --- GEMM: wave w covers k in [256w, 256w+256) --------------------------
    v8f acc = {};
    #pragma unroll
    for (int c = 0; c < 8; ++c){
      int kb = w*256 + c*32;
      BFrag ua;                            // A: h rows (batch) 0..15, convert f32->bf16
      {
        int kk = kb + hi*8;
        #pragma unroll
        for (int j = 0; j < 8; ++j) ua.s[j]     = f2bf(hst[ln][kk + j]);
        #pragma unroll
        for (int j = 0; j < 8; ++j) ua.s[8 + j] = f2bf(hst[ln][kk + 16 + j]);
      }
      BFrag ub;                            // B: column N0+ln of W_h^T
      {
        const unsigned short* p = &wlds[ln][kb + hi*8];
        ub.q[0] = *(const ushort4*)&p[0];
        ub.q[1] = *(const ushort4*)&p[4];
        ub.q[2] = *(const ushort4*)&p[16];
        ub.q[3] = *(const ushort4*)&p[20];
      }
      acc = wmma_bf16(ua.v, ub.v, acc);
    }

    if (w > 0) *(v8f*)&red[w-1][lane*8] = acc;
    __syncthreads();

    if (w == 0){
      acc = acc + *(const v8f*)&red[0][lane*8]
                + *(const v8f*)&red[1][lane*8]
                + *(const v8f*)&red[2][lane*8];
      #pragma unroll
      for (int r = 0; r < 8; ++r){
        int b = hi*8 + r;                  // batch row (C/D layout)
        int e = N0 + ln;                   // feature column
        size_t idx = ((size_t)t*Bb + b)*Dd + e;
        float al = alpha[idx];
        float be = beta[idx];
        float wv = wx[idx];
        float hp = hst[b][e];
        float vv = tanhf(acc[r] + wv);
        float hn = al*hp + be*vv;
        h[idx + (size_t)BD] = hn;          // h[t+1]
        float sg = 1.0f / (1.0f + __expf(-hn));
        out[idx] = hn * (hn * sg);         // h * silu(h)
      }
    }

    // --- grid-wide barrier: everyone sees h[t+1] ----------------------------
    __threadfence();
    __syncthreads();
    if (tid == 0){
      if (atomicAdd(bar_cnt, 1u) == (unsigned)nblk - 1u){
        atomicExch(bar_cnt, 0u);
        __threadfence();
        atomicAdd(bar_gen, 1u);
      } else {
        while (atomicAdd(bar_gen, 0u) < (unsigned)(t + 1)){
          __builtin_amdgcn_s_sleep(2);
        }
      }
    }
    __syncthreads();
  }
}

extern "C" void kernel_launch(void* const* d_in, const int* in_sizes, int n_in,
                              void* d_out, int out_size, void* d_ws, size_t ws_size,
                              hipStream_t stream)
{
  (void)n_in; (void)out_size; (void)ws_size;
  const float* x  = (const float*)d_in[0];
  const float* h0 = (const float*)d_in[1];
  const float* Wa = (const float*)d_in[2];
  const float* ba = (const float*)d_in[3];
  const float* Wb = (const float*)d_in[4];
  const float* bb = (const float*)d_in[5];
  const float* Wh = (const float*)d_in[6];
  const float* Wx = (const float*)d_in[7];
  const float* bx = (const float*)d_in[8];

  const int D = Dd, B = Bb;
  const int T = in_sizes[0] / (B * D);          // 2048
  const size_t TBD = (size_t)T * B * D;

  float* out = (float*)d_out;                   // (T,B,D)
  float* h   = out + TBD;                       // (T+1,B,D) — doubles as scan state

  float* alpha = (float*)d_ws;
  float* beta  = alpha + TBD;
  float* wx    = beta + TBD;
  unsigned* bar = (unsigned*)(wx + TBD);

  hipMemsetAsync(bar, 0, 2*sizeof(unsigned), stream);
  hipMemcpyAsync(h, h0, (size_t)B*D*sizeof(float), hipMemcpyDeviceToDevice, stream);

  dim3 g1(D/64, (T*B)/128);                     // 16 x 256 blocks
  proj_gemm<<<g1, 256, 0, stream>>>(x, Wa, Wb, Wx, ba, bb, bx, alpha, beta, wx);

  const int NBLK = D/16;                        // 64 persistent blocks
  rnn_scan<<<NBLK, 128, 0, stream>>>(Wh, alpha, beta, wx, out, h, bar, bar+1, T, NBLK);
}
